// HybridDental3DNet_17411797418424
// MI455X (gfx1250) — compile-verified, matrix operand order checked by reading, not run on hardware
//
#include <hip/hip_runtime.h>

typedef __attribute__((ext_vector_type(16))) _Float16 v16h;
typedef __attribute__((ext_vector_type(8)))  float    v8f;

#define B_  8
#define HW_IN 512
#define HW    128
#define D_    128

// ---------------------------------------------------------------------------
// Pass 1a: horizontal antialiased bilinear (triangle, scale 4 -> 8 taps)
// in  [8,512,512]  ->  tmp [8,512,128]   (both depth and xray)
// ---------------------------------------------------------------------------
__global__ void resize_h(const float* __restrict__ dm, const float* __restrict__ xr,
                         float* __restrict__ tD, float* __restrict__ tX) {
  int idx = blockIdx.x * blockDim.x + threadIdx.x;       // 8*512*128
  if (idx >= B_ * HW_IN * HW) return;
  int j = idx & 127;
  int y = (idx >> 7) & 511;
  int b = idx >> 16;
  const float W[8] = {1.f, 3.f, 5.f, 7.f, 7.f, 5.f, 3.f, 1.f};
  const float* rowD = dm + ((size_t)(b * HW_IN + y)) * HW_IN;
  const float* rowX = xr + ((size_t)(b * HW_IN + y)) * HW_IN;
  int s = 4 * j - 2;
  float aD = 0.f, aX = 0.f, ws = 0.f;
#pragma unroll
  for (int t = 0; t < 8; ++t) {
    int c = s + t;
    if (c >= 0 && c < HW_IN) {
      float w = W[t];
      aD += w * rowD[c];
      aX += w * rowX[c];
      ws += w;
    }
  }
  float inv = 1.0f / ws;                                  // edge renormalization
  tD[idx] = aD * inv;
  tX[idx] = aX * inv;
}

// ---------------------------------------------------------------------------
// Pass 1b: vertical filter + depth index
// tmp [8,512,128] -> xray_r [8,128,128] (f32), didx [8,128,128] (i32)
// ---------------------------------------------------------------------------
__global__ void resize_v(const float* __restrict__ tD, const float* __restrict__ tX,
                         float* __restrict__ xrs, int* __restrict__ didx) {
  int idx = blockIdx.x * blockDim.x + threadIdx.x;       // 8*128*128
  if (idx >= B_ * HW * HW) return;
  int j = idx & 127;
  int i = (idx >> 7) & 127;
  int b = idx >> 14;
  const float W[8] = {1.f, 3.f, 5.f, 7.f, 7.f, 5.f, 3.f, 1.f};
  int s = 4 * i - 2;
  float aD = 0.f, aX = 0.f, ws = 0.f;
#pragma unroll
  for (int t = 0; t < 8; ++t) {
    int r = s + t;
    if (r >= 0 && r < HW_IN) {
      float w = W[t];
      size_t o = ((size_t)(b * HW_IN + r)) * HW + j;
      aD += w * tD[o];
      aX += w * tX[o];
      ws += w;
    }
  }
  float inv = 1.0f / ws;
  float d = aD * inv;
  float x = aX * inv;
  int di = (int)((d / 100.0f) * 127.0f);                  // trunc (d >= 0)
  di = di < 0 ? 0 : (di > 127 ? 127 : di);
  xrs[idx]  = x;
  didx[idx] = di;
}

// ---------------------------------------------------------------------------
// Pass 2: fused gaussian depth splat + 3x3x3 box pool (/27)
// Block = 256 threads = one 16x16 (h,w) tile of one batch image.
// LDS:  S[128][256] f32 column accumulator (bank-conflict-free layout),
//       18x18 halo of (xray_r, d_idx).
// Spatial 3x3 folded into the 9-neighbor splat; depth 3-tap box expressed as
// OUT = T * S  (T = tridiagonal(1/27)) via v_wmma_f32_16x16x32_f16.
// ---------------------------------------------------------------------------
__global__ void splat_pool(const float* __restrict__ xrs, const int* __restrict__ didx,
                           float* __restrict__ out) {
  extern __shared__ float smem[];
  float* S  = smem;                    // 128*256 floats = 128 KB
  float* sx = smem + D_ * 256;         // 324 floats
  int*   sd = (int*)(sx + 324);        // 324 ints

  const int tid = threadIdx.x;
  const int b  = blockIdx.z;
  const int h0 = blockIdx.y * 16;
  const int w0 = blockIdx.x * 16;

  // zero column accumulator
  for (int i = tid; i < D_ * 256; i += 256) S[i] = 0.0f;

  // load 18x18 halo (zero outside image: 3x3 pool pads with zero)
  for (int i = tid; i < 324; i += 256) {
    int hy = h0 + i / 18 - 1;
    int wx = w0 + i % 18 - 1;
    float x = 0.0f; int d0 = 0;
    if (hy >= 0 && hy < HW && wx >= 0 && wx < HW) {
      int g = (b * HW + hy) * HW + wx;
      x = xrs[g]; d0 = didx[g];
    }
    sx[i] = x; sd[i] = d0;
  }
  __syncthreads();

  // 9-neighbor gaussian splat into own column S[:, tid]
  const float G[5] = {0.13533528f, 0.60653066f, 1.0f, 0.60653066f, 0.13533528f};
  const int hl = tid >> 4, wl = tid & 15;
  for (int dh = 0; dh < 3; ++dh) {
    for (int dw = 0; dw < 3; ++dw) {
      int n = (hl + dh) * 18 + (wl + dw);
      float x = sx[n];
      int d0 = sd[n];
#pragma unroll
      for (int k = 0; k < 5; ++k) {
        int t = d0 + k - 2;
        if (t >= 0 && t < D_) S[t * 256 + tid] += x * G[k];
      }
    }
  }
  __syncthreads();

  // Depth 3-tap box via WMMA: OUT(128x256) = T(128x128, tridiag 1/27) * S
  // 8 waves x 16 tiles each; tile = (di: 16 depth rows, cj: 16 columns).
  const int wave = tid >> 5;
  const int lane = tid & 31;
  const int laneM = lane & 15;
  const int hiHalf = lane >> 4;                 // K +8 group for lanes 16..31

  for (int tl = 0; tl < 16; ++tl) {
    int tile = wave * 16 + tl;
    int dit = tile >> 4;
    int cj  = tile & 15;
    int rowBase = dit * 16;
    int lo = rowBase - 1;  if (lo < 0)   lo = 0;
    int hi = rowBase + 16; if (hi > 127) hi = 127;

    v8f acc = {0.f, 0.f, 0.f, 0.f, 0.f, 0.f, 0.f, 0.f};
    for (int kc = (lo >> 5); kc <= (hi >> 5); ++kc) {     // 1..2 K-chunks
      v16h a, bm;
#pragma unroll
      for (int e = 0; e < 16; ++e) {
        int j = e >> 1;
        // 16-bit A 16x32 layout: VGPR j<4: K=2j,2j+1 (+8 hi-lanes); j>=4: K=16+2(j-4)...
        int kb = (j < 4 ? 2 * j : 16 + 2 * (j - 4)) + (hiHalf ? 8 : 0) + (e & 1);
        int col = kc * 32 + kb;
        int row = rowBase + laneM;
        int diff = row - col;
        a[e] = (diff >= -1 && diff <= 1) ? (_Float16)(1.0f / 27.0f) : (_Float16)0.0f;
        bm[e] = (_Float16)S[col * 256 + (cj * 16 + laneM)];
      }
      acc = __builtin_amdgcn_wmma_f32_16x16x32_f16(
          /*neg_a=*/false, a, /*neg_b=*/false, bm,
          /*c_mod=*/(short)0, acc, /*reuse_a=*/false, /*reuse_b=*/false);
    }

    // D layout: lane<16 -> M=r, lane>=16 -> M=r+8 ; N = laneM
    int dBase = rowBase + (hiHalf ? 8 : 0);
    int h = h0 + cj, w = w0 + laneM;
    float* op = out + (((size_t)(b * D_ + dBase) * HW + h) * HW + w);
#pragma unroll
    for (int r = 0; r < 8; ++r) op[(size_t)r * HW * HW] = acc[r];
  }
}

// ---------------------------------------------------------------------------
extern "C" void kernel_launch(void* const* d_in, const int* in_sizes, int n_in,
                              void* d_out, int out_size, void* d_ws, size_t ws_size,
                              hipStream_t stream) {
  const float* depth = (const float*)d_in[0];
  const float* xray  = (const float*)d_in[1];
  float* ws  = (float*)d_ws;
  float* tD  = ws;                       // 8*512*128 = 524288 f32
  float* tX  = ws + 524288;              // 524288 f32
  float* xrs = ws + 1048576;             // 131072 f32
  int*   dix = (int*)(ws + 1179648);     // 131072 i32   (total ~5.2 MB scratch)
  float* out = (float*)d_out;

  resize_h<<<(B_ * HW_IN * HW + 255) / 256, 256, 0, stream>>>(depth, xray, tD, tX);
  resize_v<<<(B_ * HW * HW + 255) / 256, 256, 0, stream>>>(tD, tX, xrs, dix);

  dim3 grid(HW / 16, HW / 16, B_);       // 8 x 8 x 8 blocks
  size_t smem = (size_t)D_ * 256 * sizeof(float) + 324 * sizeof(float) + 324 * sizeof(int);
  splat_pool<<<grid, 256, smem, stream>>>(xrs, dix, out);
}
